// HeteroGNN_25812753449244
// MI455X (gfx1250) — matrix operand order repaired and checked by reading, not compiled, not run on hardware
//
#include <hip/hip_runtime.h>

#define CDIV(a, b) (((a) + (b) - 1) / (b))

typedef __attribute__((ext_vector_type(16))) __bf16        v16bf;
typedef __attribute__((ext_vector_type(8)))  float          v8f;
typedef __attribute__((ext_vector_type(8)))  unsigned short v8u16;
typedef __attribute__((ext_vector_type(16))) unsigned short v16u16;

__device__ __forceinline__ unsigned short f2bf(float f) {
  unsigned u = __float_as_uint(f);
  u += 0x7fffu + ((u >> 16) & 1u);   // round to nearest even
  return (unsigned short)(u >> 16);
}

// ---------------- prep kernels ----------------

// Pack W [K, Nout] f32 row-major into CDNA5 WMMA B-fragment layout (bf16):
// P[((tn*(K/32)+step)*32 + lane)*16 + i], k = step*32 + (lane/16)*8 + (i%8) + (i/8)*16, n = tn*16 + lane%16
__global__ void pack_b_k(const float* __restrict__ W, unsigned short* __restrict__ P,
                         int K, int Nout) {
  int idx = blockIdx.x * blockDim.x + threadIdx.x;
  int total = K * Nout;
  if (idx >= total) return;
  int i    = idx & 15;
  int lane = (idx >> 4) & 31;
  int rest = idx >> 9;
  int ksteps = K >> 5;
  int step = rest % ksteps;
  int tn   = rest / ksteps;
  int k = step * 32 + ((lane >> 4) << 3) + (i & 7) + ((i >> 3) << 4);
  int n = tn * 16 + (lane & 15);
  P[idx] = f2bf(W[(size_t)k * Nout + n]);
}

// out[j] = dot(W[j, 0:cols], a)   (row-dot; folds Wd@a_d and We@a_e so hd/he are never built)
__global__ void matvec_rows_k(const float* __restrict__ W, const float* __restrict__ a,
                              float* __restrict__ out, int rows, int cols) {
  int j = blockIdx.x * blockDim.x + threadIdx.x;
  if (j >= rows) return;
  float s = 0.f;
  for (int c = 0; c < cols; ++c) s += W[(size_t)j * cols + c] * a[c];
  out[j] = s;
}

// vectorized fp32 -> bf16 stream (float4 in, ushort4 out)
__global__ void f32_to_bf16_k(const float* __restrict__ X, unsigned short* __restrict__ Y,
                              int n4) {
  int i = blockIdx.x * blockDim.x + threadIdx.x;
  if (i >= n4) return;
  float4 x = ((const float4*)X)[i];
  ushort4 y;
  y.x = f2bf(x.x); y.y = f2bf(x.y); y.z = f2bf(x.z); y.w = f2bf(x.w);
  ((ushort4*)Y)[i] = y;
}

__global__ void init_conv_k(float* __restrict__ acc, int* __restrict__ Mkey,
                            float* __restrict__ den, int Nn) {
  int i = blockIdx.x * blockDim.x + threadIdx.x;
  if (i < Nn * 128) acc[i] = 0.f;
  if (i < Nn) { Mkey[i] = (int)0x80000000; den[i] = 0.f; }
}

// ------- WMMA GEMM: C[M,Nout] = bf16(A[M,128]) @ Bpacked (+bias), 16x64 tile per wave -------

__global__ void __launch_bounds__(256)
gemm_bf16_wmma_k(const unsigned short* __restrict__ A,   // [M,128] bf16 row-major
                 const unsigned short* __restrict__ Bp,  // packed fragments
                 const float* __restrict__ bias,         // [Nout] or nullptr
                 float* __restrict__ Cm, int M, int Nout) {
  const int lane = threadIdx.x & 31;
  const int wave = (blockIdx.x * blockDim.x + threadIdx.x) >> 5;
  const int nTG  = Nout >> 6;                 // groups of 4 N-tiles (Nout = 64 or 128)
  if (wave >= (M >> 4) * nTG) return;         // wave-uniform: EXEC stays all-1 for WMMA
  const int tm = wave / nTG;
  const int tg = wave % nTG;
  const int m = (tm << 4) + (lane & 15);
  const int ahalf = (lane >> 4) << 3;         // A: lanes 16-31 hold K-offsets +8 in each 16-run
  const unsigned short* arow  = A + (size_t)m * 128;
  const unsigned short* bbase = Bp + (size_t)(tg * 4) * 2048 + (size_t)lane * 16;
  v8f c0 = {}, c1 = {}, c2 = {}, c3 = {};
#pragma unroll
  for (int s = 0; s < 4; ++s) {
    const int kb = s * 32 + ahalf;
    v8u16 a0 = *(const v8u16*)(arow + kb);        // K = kb+0..7
    v8u16 a1 = *(const v8u16*)(arow + kb + 16);   // K = kb+16..23
    v16u16 au = __builtin_shufflevector(a0, a1, 0,1,2,3,4,5,6,7,8,9,10,11,12,13,14,15);
    v16bf av = __builtin_bit_cast(v16bf, au);
    const unsigned short* bs = bbase + (size_t)s * 512;
    v16bf b0 = __builtin_bit_cast(v16bf, *(const v16u16*)(bs + 0 * 2048));
    v16bf b1 = __builtin_bit_cast(v16bf, *(const v16u16*)(bs + 1 * 2048));
    v16bf b2 = __builtin_bit_cast(v16bf, *(const v16u16*)(bs + 2 * 2048));
    v16bf b3 = __builtin_bit_cast(v16bf, *(const v16u16*)(bs + 3 * 2048));
    c0 = __builtin_amdgcn_wmma_f32_16x16x32_bf16(false, av, false, b0, (short)0, c0, false, false);
    c1 = __builtin_amdgcn_wmma_f32_16x16x32_bf16(false, av, false, b1, (short)0, c1, false, false);
    c2 = __builtin_amdgcn_wmma_f32_16x16x32_bf16(false, av, false, b2, (short)0, c2, false, false);
    c3 = __builtin_amdgcn_wmma_f32_16x16x32_bf16(false, av, false, b3, (short)0, c3, false, false);
  }
  const int rbase = (tm << 4) + ((lane >> 4) << 3);
  v8f cc[4] = {c0, c1, c2, c3};
#pragma unroll
  for (int t = 0; t < 4; ++t) {
    const int n = (tg * 4 + t) * 16 + (lane & 15);
    const float b = bias ? bias[n] : 0.f;
#pragma unroll
    for (int r = 0; r < 8; ++r)
      Cm[(size_t)(rbase + r) * Nout + n] = cc[t][r] + b;
  }
}

// ---------------- attention kernels ----------------

// out[n] = dot(X[n, 0:128], v) ; one wave per row, float4 per lane + butterfly reduce
__global__ void rowdot_k(const float* __restrict__ X, const float* __restrict__ v,
                         float* __restrict__ out, int Nrows) {
  int wave = (blockIdx.x * blockDim.x + threadIdx.x) >> 5;
  int lane = threadIdx.x & 31;
  if (wave >= Nrows) return;
  float4 x  = *(const float4*)(X + (size_t)wave * 128 + lane * 4);
  float4 vv = *(const float4*)(v + lane * 4);
  float s = x.x * vv.x + x.y * vv.y + x.z * vv.z + x.w * vv.w;
#pragma unroll
  for (int o = 16; o > 0; o >>= 1) s += __shfl_down(s, o, 32);
  if (lane == 0) out[wave] = s;
}

__device__ __forceinline__ int fkey(float f) {   // order-preserving float->int (involution)
  int b = __float_as_int(f);
  return b ^ ((b >> 31) & 0x7fffffff);
}

__global__ void edge_alpha_k(const float* __restrict__ ea, const float* __restrict__ wev,
                             const int* __restrict__ row, const int* __restrict__ col,
                             const float* __restrict__ asrc, const float* __restrict__ adst,
                             float* __restrict__ alpha, int* __restrict__ Mkey, int E) {
  int e = blockIdx.x * blockDim.x + threadIdx.x;
  if (e >= E) return;
  const float4* ev = (const float4*)(ea + (size_t)e * 16);
  const float4* wv = (const float4*)wev;
  float ae = 0.f;
#pragma unroll
  for (int j = 0; j < 4; ++j) {
    float4 a = ev[j], b = wv[j];
    ae += a.x * b.x + a.y * b.y + a.z * b.z + a.w * b.w;
  }
  int c = col[e];
  float al = asrc[row[e]] + adst[c] + ae;
  al = (al > 0.f) ? al : 0.2f * al;            // leaky_relu(0.2)
  alpha[e] = al;
  atomicMax(&Mkey[c], fkey(al));
}

__global__ void edge_exp_k(const int* __restrict__ Mkey, const int* __restrict__ col,
                           float* __restrict__ alpha, float* __restrict__ den, int E) {
  int e = blockIdx.x * blockDim.x + threadIdx.x;
  if (e >= E) return;
  int c = col[e];
  int k = Mkey[c];
  float m = __int_as_float(k ^ ((k >> 31) & 0x7fffffff));
  float ex = __expf(alpha[e] - m);
  alpha[e] = ex;
  atomicAdd(&den[c], ex);
}

// one wave per edge: acc[col,:] += (ex/den[col]) * hs[row,:]
__global__ void edge_scatter_k(const float* __restrict__ hs, const float* __restrict__ ex,
                               const float* __restrict__ den, const int* __restrict__ row,
                               const int* __restrict__ col, float* __restrict__ acc, int E) {
  int wave = (blockIdx.x * blockDim.x + threadIdx.x) >> 5;
  int lane = threadIdx.x & 31;
  if (wave >= E) return;
  int r = row[wave], c = col[wave];
  float w = ex[wave] / den[c];
  float4 h = *(const float4*)(hs + (size_t)r * 128 + lane * 4);
  float* dst = acc + (size_t)c * 128 + lane * 4;
  atomicAdd(dst + 0, w * h.x);
  atomicAdd(dst + 1, w * h.y);
  atomicAdd(dst + 2, w * h.z);
  atomicAdd(dst + 3, w * h.w);
}

__global__ void finalize_relu_k(const float* __restrict__ acc, const float* __restrict__ bias,
                                float* __restrict__ xout, int total) {
  int i = blockIdx.x * blockDim.x + threadIdx.x;
  if (i >= total) return;
  float v = acc[i] + bias[i & 127];
  xout[i] = v > 0.f ? v : 0.f;
}

// ---------------- host orchestration ----------------

extern "C" void kernel_launch(void* const* d_in, const int* in_sizes, int n_in,
                              void* d_out, int out_size, void* d_ws, size_t ws_size,
                              hipStream_t stream) {
  (void)n_in; (void)out_size; (void)ws_size;
  const float* x_user    = (const float*)d_in[0];
  const float* x_item    = (const float*)d_in[1];
  const float* ea_ui     = (const float*)d_in[2];
  const float* ea_iu     = (const float*)d_in[3];
  const float* W_src     = (const float*)d_in[4];
  const float* W_dst     = (const float*)d_in[5];
  const float* W_edge    = (const float*)d_in[6];
  const float* att_src   = (const float*)d_in[7];
  const float* att_dst   = (const float*)d_in[8];
  const float* att_edge  = (const float*)d_in[9];
  const float* conv_bias = (const float*)d_in[10];
  const float* lin0_W    = (const float*)d_in[11];
  const float* lin0_b    = (const float*)d_in[12];
  const float* lin1_W    = (const float*)d_in[13];
  const float* lin1_b    = (const float*)d_in[14];
  const int*   ei_ui     = (const int*)d_in[15];
  const int*   ei_iu     = (const int*)d_in[16];

  const int Nn = in_sizes[0] / 128;      // 100000
  const int Ee = in_sizes[15] / 2;       // 1600000
  const size_t NC = (size_t)Nn * 128;

  size_t off = 0;
  auto carve = [&](size_t bytes) -> void* {
    void* p = (char*)d_ws + off;
    off += (bytes + 255) & ~(size_t)255;
    return p;
  };
  float* xcur_u = (float*)carve(NC * 4);
  float* xcur_i = (float*)carve(NC * 4);
  float* acc_u  = (float*)carve(NC * 4);
  float* acc_i  = (float*)carve(NC * 4);
  float* hs     = (float*)carve(NC * 4);            // also reused as y0 in the MLP head
  unsigned short* xbf = (unsigned short*)carve(NC * 2);
  float* asrc   = (float*)carve((size_t)Nn * 4);
  float* adst   = (float*)carve((size_t)Nn * 4);
  int*   Mkey   = (int*)carve((size_t)Nn * 4);
  float* den    = (float*)carve((size_t)Nn * 4);
  float* ealpha = (float*)carve((size_t)Ee * 4);
  unsigned short* Wpack  = (unsigned short*)carve((size_t)4 * 16384 * 2);
  unsigned short* l0pack = (unsigned short*)carve((size_t)2 * 16384 * 2);
  unsigned short* l1pack = (unsigned short*)carve((size_t)2 * 8192 * 2);
  float* wdvec = (float*)carve((size_t)4 * 128 * 4);
  float* wevec = (float*)carve((size_t)4 * 16 * 4);

  // ---- prep: pack weights to WMMA-B layout; fold attention vectors ----
  for (int lt = 0; lt < 4; ++lt) {
    pack_b_k<<<CDIV(128 * 128, 256), 256, 0, stream>>>(
        W_src + (size_t)lt * 16384, Wpack + (size_t)lt * 16384, 128, 128);
    matvec_rows_k<<<2, 64, 0, stream>>>(
        W_dst + (size_t)lt * 16384, att_dst + lt * 128, wdvec + lt * 128, 128, 128);
    matvec_rows_k<<<1, 64, 0, stream>>>(
        W_edge + (size_t)lt * 2048, att_edge + lt * 128, wevec + lt * 16, 16, 128);
  }
  for (int i = 0; i < 2; ++i) {
    pack_b_k<<<CDIV(16384, 256), 256, 0, stream>>>(
        lin0_W + (size_t)i * 16384, l0pack + (size_t)i * 16384, 128, 128);
    pack_b_k<<<CDIV(8192, 256), 256, 0, stream>>>(
        lin1_W + (size_t)i * 8192, l1pack + (size_t)i * 8192, 128, 64);
  }

  const int cvt_blocks    = CDIV((int)(NC / 4), 256);
  const int gemm128_blocks = CDIV((Nn / 16) * 2 * 32, 256);  // nTG=2 for Nout=128
  const int gemm64_blocks  = CDIV((Nn / 16) * 1 * 32, 256);  // nTG=1 for Nout=64

  auto run_conv = [&](int l, int t, const float* xsrc, const float* xdst,
                      const float* ea, const int* ei, float* acc) {
    const int lt = l * 2 + t;
    f32_to_bf16_k<<<cvt_blocks, 256, 0, stream>>>(xsrc, xbf, (int)(NC / 4));
    init_conv_k<<<CDIV((int)NC, 256), 256, 0, stream>>>(acc, Mkey, den, Nn);
    gemm_bf16_wmma_k<<<gemm128_blocks, 256, 0, stream>>>(
        xbf, Wpack + (size_t)lt * 16384, nullptr, hs, Nn, 128);
    rowdot_k<<<CDIV(Nn, 8), 256, 0, stream>>>(hs, att_src + lt * 128, asrc, Nn);
    rowdot_k<<<CDIV(Nn, 8), 256, 0, stream>>>(xdst, wdvec + lt * 128, adst, Nn);
    edge_alpha_k<<<CDIV(Ee, 256), 256, 0, stream>>>(
        ea, wevec + lt * 16, ei, ei + Ee, asrc, adst, ealpha, Mkey, Ee);
    edge_exp_k<<<CDIV(Ee, 256), 256, 0, stream>>>(Mkey, ei + Ee, ealpha, den, Ee);
    edge_scatter_k<<<CDIV(Ee, 8), 256, 0, stream>>>(hs, ealpha, den, ei, ei + Ee, acc, Ee);
  };

  const float* cu = x_user;
  const float* ci = x_item;
  for (int l = 0; l < 2; ++l) {
    run_conv(l, 0, cu, ci, ea_ui, ei_ui, acc_i);   // user -> item
    run_conv(l, 1, ci, cu, ea_iu, ei_iu, acc_u);   // item -> user
    finalize_relu_k<<<CDIV((int)NC, 256), 256, 0, stream>>>(
        acc_i, conv_bias + (size_t)(l * 2 + 0) * 128, xcur_i, (int)NC);
    finalize_relu_k<<<CDIV((int)NC, 256), 256, 0, stream>>>(
        acc_u, conv_bias + (size_t)(l * 2 + 1) * 128, xcur_u, (int)NC);
    cu = xcur_u; ci = xcur_i;
  }

  // ---- MLP head: (x @ lin0 + b0) @ lin1 + b1, per node type ----
  float* outp = (float*)d_out;
  const float* xs[2] = {cu, ci};
  for (int i = 0; i < 2; ++i) {
    f32_to_bf16_k<<<cvt_blocks, 256, 0, stream>>>(xs[i], xbf, (int)(NC / 4));
    gemm_bf16_wmma_k<<<gemm128_blocks, 256, 0, stream>>>(
        xbf, l0pack + (size_t)i * 16384, lin0_b + (size_t)i * 128, hs, Nn, 128);
    f32_to_bf16_k<<<cvt_blocks, 256, 0, stream>>>(hs, xbf, (int)(NC / 4));
    gemm_bf16_wmma_k<<<gemm64_blocks, 256, 0, stream>>>(
        xbf, l1pack + (size_t)i * 8192, lin1_b + (size_t)i * 64,
        outp + (size_t)i * Nn * 64, Nn, 64);
  }
}
